// BlockWithReturnAttention_1846835938111
// MI455X (gfx1250) — compile-verified, matrix operand order checked by reading, not run on hardware
//
#include <hip/hip_runtime.h>
#include <hip/hip_bf16.h>
#include <math.h>

// ---------------------------------------------------------------------------
// Types for CDNA5 WMMA (wave32): v_wmma_f32_16x16x32_bf16
// ---------------------------------------------------------------------------
typedef __attribute__((ext_vector_type(16))) __bf16 v16bf;
typedef __attribute__((ext_vector_type(8)))  __bf16 bf16x8;
typedef __attribute__((ext_vector_type(4)))  __bf16 bf16x4;
typedef __attribute__((ext_vector_type(8)))  float  v8f;

__device__ __forceinline__ v8f wmma_bf16(v16bf a, v16bf b, v8f c) {
  // 8 args: (neg_a, A, neg_b, B, c_mod, C, reuse_a, reuse_b)
  return __builtin_amdgcn_wmma_f32_16x16x32_bf16(
      false, a, false, b, (short)0, c, false, false);
}

__device__ __forceinline__ v8f zero8() {
  v8f z;
#pragma unroll
  for (int i = 0; i < 8; ++i) z[i] = 0.f;
  return z;
}

// ---------------------------------------------------------------------------
// Async global->LDS staging (CDNA5 GLOBAL_LOAD_ASYNC_TO_LDS_B128, ASYNCcnt).
// Builtin signature (from toolchain diagnostic): param0 is
// int4 __attribute__((address_space(1)))* (global), param1 is AS(3) (LDS).
// Guarded: falls back to VGPR round-trip copies when the builtin is absent.
// ---------------------------------------------------------------------------
#if __has_builtin(__builtin_amdgcn_global_load_async_to_lds_b128)
#define ASYNC_LDS 1
typedef int v4i_ __attribute__((vector_size(16)));
__device__ __forceinline__ void async_b128(const __bf16* g, __bf16* l) {
  __builtin_amdgcn_global_load_async_to_lds_b128(
      (__attribute__((address_space(1))) v4i_*)g,
      (__attribute__((address_space(3))) v4i_*)l, 0, 0);
}
#endif

__device__ __forceinline__ void wait_async0() {
#ifdef ASYNC_LDS
#if __has_builtin(__builtin_amdgcn_s_wait_asynccnt)
  __builtin_amdgcn_s_wait_asynccnt(0);
#else
  asm volatile("s_wait_asynccnt 0x0" ::: "memory");
#endif
#endif
}

// A/B fragment loader from a row-major LDS tile (row stride `ld` elements,
// ld % 8 == 0 so the two 16B chunks are aligned).
// 16-bit A layout: lanes 0-15 -> K 0..7 and 16..23 of row (lane&15),
//                  lanes 16-31 -> K 8..15 and 24..31 of the same row.
__device__ __forceinline__ v16bf load_frag(const __bf16* base, int ld, int lane) {
  const int hl = (lane >> 4) & 1;
  const int r  = lane & 15;
  const __bf16* p = base + r * ld + hl * 8;
  bf16x8 lo = *(const bf16x8*)(p);
  bf16x8 hi = *(const bf16x8*)(p + 16);
  v16bf f;
#pragma unroll
  for (int i = 0; i < 8; ++i) { f[i] = lo[i]; f[i + 8] = hi[i]; }
  return f;
}

// ---------------------------------------------------------------------------
// fp32 -> bf16 conversion (weights), 4 elements / thread
// ---------------------------------------------------------------------------
__global__ __launch_bounds__(256) void cvt_f32_bf16_kernel(
    const float* __restrict__ in, __bf16* __restrict__ out) {
  size_t i = (size_t)blockIdx.x * 256 + threadIdx.x;
  float4 v = ((const float4*)in)[i];
  bf16x4 o;
  o[0] = (__bf16)v.x; o[1] = (__bf16)v.y; o[2] = (__bf16)v.z; o[3] = (__bf16)v.w;
  ((bf16x4*)out)[i] = o;
}

// ---------------------------------------------------------------------------
// LayerNorm: one 256-thread block per row of 1024, bf16 output
// ---------------------------------------------------------------------------
__global__ __launch_bounds__(256) void ln_kernel(
    const float* __restrict__ x, const float* __restrict__ g,
    const float* __restrict__ bta, __bf16* __restrict__ y) {
  __shared__ float sa[256];
  __shared__ float sb[256];
  const int tid = threadIdx.x;
  const size_t row = blockIdx.x;
  const float4 v = ((const float4*)(x + row * 1024))[tid];
  float s = v.x + v.y + v.z + v.w;
  float q = v.x * v.x + v.y * v.y + v.z * v.z + v.w * v.w;
  sa[tid] = s; sb[tid] = q;
  __syncthreads();
  for (int st = 128; st > 0; st >>= 1) {
    if (tid < st) { sa[tid] += sa[tid + st]; sb[tid] += sb[tid + st]; }
    __syncthreads();
  }
  const float mu  = sa[0] * (1.f / 1024.f);
  const float var = sb[0] * (1.f / 1024.f) - mu * mu;
  const float rs  = rsqrtf(var + 1e-5f);
  const int c = tid * 4;
  bf16x4 o;
  o[0] = (__bf16)((v.x - mu) * rs * g[c + 0] + bta[c + 0]);
  o[1] = (__bf16)((v.y - mu) * rs * g[c + 1] + bta[c + 1]);
  o[2] = (__bf16)((v.z - mu) * rs * g[c + 2] + bta[c + 2]);
  o[3] = (__bf16)((v.w - mu) * rs * g[c + 3] + bta[c + 3]);
  *(bf16x4*)&y[row * 1024 + c] = o;
}

// ---------------------------------------------------------------------------
// WMMA GEMM: C[M,N] = A[M,K](bf16) * W[K,N](bf16) with fused epilogue.
// Block tile 128x128, K-step 64 staged in LDS. 8 waves, each owns 64x32 out.
// A tile staged with async global->LDS; W tile transposed via VGPRs while the
// async engine streams A; next W stage prefetched into L2.
// ---------------------------------------------------------------------------
template <bool BIAS, bool GELU_, bool RESID, bool OUTBF>
__global__ __launch_bounds__(256) void gemm_bf16_kernel(
    const __bf16* __restrict__ A, const __bf16* __restrict__ W,
    float* __restrict__ outf, __bf16* __restrict__ outb,
    const float* __restrict__ bias, const float* __restrict__ resid,
    int M, int N, int K) {
  __shared__ __align__(16) __bf16 a_s[128 * 72];
  __shared__ __align__(16) __bf16 b_s[128 * 72];

  const int tid  = threadIdx.x;
  const int lane = tid & 31;
  const int wave = tid >> 5;
  const int wm = wave & 1;   // 2 wave-rows  (64 rows each)
  const int wn = wave >> 1;  // 4 wave-cols  (32 cols each)
  const int m0 = blockIdx.y * 128;
  const int n0 = blockIdx.x * 128;

  v8f acc[4][2];
#pragma unroll
  for (int a = 0; a < 4; ++a)
#pragma unroll
    for (int b = 0; b < 2; ++b) acc[a][b] = zero8();

  const int ncp = tid & 127;          // copy helper: column inside B tile
  const int kcp = (tid >> 7) * 32;    // copy helper: K half

  for (int k0 = 0; k0 < K; k0 += 64) {
    __syncthreads();
    // A tile: 128 rows x 64 K
#ifdef ASYNC_LDS
#pragma unroll
    for (int c = tid; c < 128 * 8; c += 256) {      // 16B chunks
      const int r  = c >> 3;
      const int kk = (c & 7) << 3;
      async_b128(&A[(size_t)(m0 + r) * K + k0 + kk], &a_s[r * 72 + kk]);
    }
#else
#pragma unroll
    for (int c = tid; c < 128 * 16; c += 256) {     // 8B chunks
      const int r  = c >> 4;
      const int kk = (c & 15) << 2;
      *(bf16x4*)&a_s[r * 72 + kk] =
          *(const bf16x4*)&A[(size_t)(m0 + r) * K + k0 + kk];
    }
#endif
    // Prefetch next W stage into cache while this one is consumed.
    if (k0 + 64 < K)
      __builtin_prefetch(&W[(size_t)(k0 + 64 + kcp) * N + n0 + ncp], 0, 0);
    // W tile, stored transposed in LDS: b_s[n][k] = W[k0+k][n0+n]
#pragma unroll 8
    for (int kk = 0; kk < 32; ++kk) {
      b_s[ncp * 72 + kcp + kk] = W[(size_t)(k0 + kcp + kk) * N + n0 + ncp];
    }
    wait_async0();
    __syncthreads();

#pragma unroll
    for (int kk = 0; kk < 64; kk += 32) {
      v16bf bf0 = load_frag(&b_s[(wn * 32 + 0)  * 72 + kk], 72, lane);
      v16bf bf1 = load_frag(&b_s[(wn * 32 + 16) * 72 + kk], 72, lane);
#pragma unroll
      for (int tm = 0; tm < 4; ++tm) {
        v16bf af = load_frag(&a_s[(wm * 64 + tm * 16) * 72 + kk], 72, lane);
        acc[tm][0] = wmma_bf16(af, bf0, acc[tm][0]);
        acc[tm][1] = wmma_bf16(af, bf1, acc[tm][1]);
      }
    }
  }

  // Epilogue. C layout: VGPR i -> row (i + 8*half), col = lane&15.
  const int hl = (lane >> 4) & 1;
  const int cl = lane & 15;
#pragma unroll
  for (int tm = 0; tm < 4; ++tm)
#pragma unroll
    for (int tn = 0; tn < 2; ++tn)
#pragma unroll
      for (int i = 0; i < 8; ++i) {
        const int row = m0 + wm * 64 + tm * 16 + i + hl * 8;
        const int col = n0 + wn * 32 + tn * 16 + cl;
        float v = acc[tm][tn][i];
        if (BIAS)  v += bias[col];
        if (GELU_) v = 0.5f * v * (1.0f + erff(v * 0.70710678118654752f));
        if (RESID) v += resid[(size_t)row * N + col];
        if (OUTBF) outb[(size_t)row * N + col] = (__bf16)v;
        else       outf[(size_t)row * N + col] = v;
      }
}

// ---------------------------------------------------------------------------
// Flash attention: block = (b, h, 128 q-rows). qkv is bf16 [8192, 3072]
// (q | k | v each 1024 wide, head h at cols h*64). Output bf16 [8192, 1024]
// laid out [B, N, H*Dh] ready for the proj GEMM.
// ---------------------------------------------------------------------------
__global__ __launch_bounds__(256) void attn_kernel(
    const __bf16* __restrict__ qkv, __bf16* __restrict__ out) {
  __shared__ __align__(16) __bf16 q_s[128 * 72];       // Q rows x Dh
  __shared__ __align__(16) __bf16 k_s[64 * 72];        // keys  x Dh
  __shared__ __align__(16) __bf16 vt_s[64 * 72];       // Dh x keys (V^T)
  __shared__ __align__(16) __bf16 p_s[8 * 16 * 72];    // per-wave P tile

  const int tid  = threadIdx.x;
  const int lane = tid & 31;
  const int wave = tid >> 5;
  const int hl = (lane >> 4) & 1;
  const int cl = lane & 15;

  const int blk = blockIdx.x;
  const int qb  = blk & 7;
  const int bh  = blk >> 3;
  const int b   = bh >> 4;
  const int h   = bh & 15;
  const int q0  = qb * 128;
  const size_t rowbase = (size_t)b * 1024;
  const int qoff = h * 64;

  // Load Q tile (stable across key loop)
#ifdef ASYNC_LDS
  for (int c = tid; c < 128 * 8; c += 256) {
    const int r = c >> 3, dd = (c & 7) << 3;
    async_b128(&qkv[(rowbase + q0 + r) * 3072 + qoff + dd], &q_s[r * 72 + dd]);
  }
#else
  for (int c = tid; c < 128 * 16; c += 256) {
    const int r = c >> 4, dd = (c & 15) << 2;
    *(bf16x4*)&q_s[r * 72 + dd] =
        *(const bf16x4*)&qkv[(rowbase + q0 + r) * 3072 + qoff + dd];
  }
#endif

  v8f o[4];
#pragma unroll
  for (int td = 0; td < 4; ++td) o[td] = zero8();
  float mrow[8], lrow[8];
#pragma unroll
  for (int i = 0; i < 8; ++i) { mrow[i] = -1e30f; lrow[i] = 0.f; }

  const int dcp = tid & 63;
  const int jb  = (tid >> 6) * 16;
  const int r0  = wave * 16;
  __bf16* pw = &p_s[wave * 16 * 72];

  for (int kt = 0; kt < 16; ++kt) {
    __syncthreads();
    // K tile: [key][Dh]  (async engine streams it while we transpose V)
#ifdef ASYNC_LDS
    for (int c = tid; c < 64 * 8; c += 256) {
      const int r = c >> 3, dd = (c & 7) << 3;
      async_b128(&qkv[(rowbase + kt * 64 + r) * 3072 + 1024 + qoff + dd],
                 &k_s[r * 72 + dd]);
    }
#else
    for (int c = tid; c < 64 * 16; c += 256) {
      const int r = c >> 4, dd = (c & 15) << 2;
      *(bf16x4*)&k_s[r * 72 + dd] =
          *(const bf16x4*)&qkv[(rowbase + kt * 64 + r) * 3072 + 1024 + qoff + dd];
    }
#endif
    // V tile transposed: vt_s[d][key]
#pragma unroll 4
    for (int jj = 0; jj < 16; ++jj) {
      const int j = jb + jj;
      vt_s[dcp * 72 + j] =
          qkv[(rowbase + kt * 64 + j) * 3072 + 2048 + qoff + dcp];
    }
    wait_async0();
    __syncthreads();

    // S = Q * K^T  (16 q-rows x 64 keys per wave)
    v8f s[4];
#pragma unroll
    for (int tn = 0; tn < 4; ++tn) s[tn] = zero8();
#pragma unroll
    for (int kk = 0; kk < 64; kk += 32) {
      v16bf aq = load_frag(&q_s[r0 * 72 + kk], 72, lane);
#pragma unroll
      for (int tn = 0; tn < 4; ++tn) {
        v16bf bk = load_frag(&k_s[tn * 16 * 72 + kk], 72, lane);
        s[tn] = wmma_bf16(aq, bk, s[tn]);
      }
    }

    // Online softmax update. Row r = i + 8*half lives across one 16-lane half.
#pragma unroll
    for (int i = 0; i < 8; ++i) {
      float v0 = s[0][i] * 0.125f, v1 = s[1][i] * 0.125f;
      float v2 = s[2][i] * 0.125f, v3 = s[3][i] * 0.125f;
      float mx = fmaxf(fmaxf(v0, v1), fmaxf(v2, v3));
#pragma unroll
      for (int off = 8; off >= 1; off >>= 1)
        mx = fmaxf(mx, __shfl_xor(mx, off, 32));
      const float mnew = fmaxf(mrow[i], mx);
      const float corr = __expf(mrow[i] - mnew);
      mrow[i] = mnew;
      const float p0 = __expf(v0 - mnew), p1 = __expf(v1 - mnew);
      const float p2 = __expf(v2 - mnew), p3 = __expf(v3 - mnew);
      s[0][i] = p0; s[1][i] = p1; s[2][i] = p2; s[3][i] = p3;
      float rs = p0 + p1 + p2 + p3;
#pragma unroll
      for (int off = 8; off >= 1; off >>= 1)
        rs += __shfl_xor(rs, off, 32);
      lrow[i] = lrow[i] * corr + rs;
#pragma unroll
      for (int td = 0; td < 4; ++td) o[td][i] *= corr;
    }

    // Spill P to per-wave LDS to re-enter the A-fragment layout.
#pragma unroll
    for (int tn = 0; tn < 4; ++tn)
#pragma unroll
      for (int i = 0; i < 8; ++i)
        pw[(i + hl * 8) * 72 + tn * 16 + cl] = (__bf16)s[tn][i];

    // O += P * V   (K dim = 64 keys)
#pragma unroll
    for (int kk = 0; kk < 64; kk += 32) {
      v16bf ap = load_frag(pw + kk, 72, lane);
#pragma unroll
      for (int td = 0; td < 4; ++td) {
        v16bf bv = load_frag(&vt_s[td * 16 * 72 + kk], 72, lane);
        o[td] = wmma_bf16(ap, bv, o[td]);
      }
    }
    __syncthreads();
  }

  // Normalize and write [B,N,H*Dh] bf16
#pragma unroll
  for (int td = 0; td < 4; ++td)
#pragma unroll
    for (int i = 0; i < 8; ++i) {
      const int row = q0 + r0 + i + hl * 8;
      out[(rowbase + row) * 1024 + qoff + td * 16 + cl] =
          (__bf16)(o[td][i] / lrow[i]);
    }
}

// ---------------------------------------------------------------------------
// Launcher
// ---------------------------------------------------------------------------
extern "C" void kernel_launch(void* const* d_in, const int* in_sizes, int n_in,
                              void* d_out, int out_size, void* d_ws,
                              size_t ws_size, hipStream_t stream) {
  (void)in_sizes; (void)n_in; (void)out_size; (void)ws_size;
  const float* x      = (const float*)d_in[0];
  const float* ln1_g  = (const float*)d_in[1];
  const float* ln1_b  = (const float*)d_in[2];
  const float* w_qkv  = (const float*)d_in[3];
  const float* w_proj = (const float*)d_in[4];
  const float* b_proj = (const float*)d_in[5];
  const float* ln2_g  = (const float*)d_in[6];
  const float* ln2_b  = (const float*)d_in[7];
  const float* w_fc1  = (const float*)d_in[8];
  const float* b_fc1  = (const float*)d_in[9];
  const float* w_fc2  = (const float*)d_in[10];
  const float* b_fc2  = (const float*)d_in[11];
  float* out = (float*)d_out;

  // Workspace layout (bytes). hbuf reuses the dead wqkv+qkv+attn region.
  char* ws = (char*)d_ws;
  __bf16* xn     = (__bf16*)(ws + 0);            // 16.78 MB (also reused as xn2)
  __bf16* wqkvb  = (__bf16*)(ws + 16777216);     //  6.29 MB
  __bf16* qkvb   = (__bf16*)(ws + 23068672);     // 50.33 MB
  __bf16* attnb  = (__bf16*)(ws + 73400320);     // 16.78 MB
  __bf16* hbuf   = (__bf16*)(ws + 16777216);     // 67.11 MB (after attn dead)
  __bf16* wprojb = (__bf16*)(ws + 90177536);     //  2.10 MB
  float*  x1     = (float*) (ws + 92274688);     // 33.55 MB
  __bf16* wfc1b  = (__bf16*)(ws + 125829120);    //  8.39 MB
  __bf16* wfc2b  = (__bf16*)(ws + 134217728);    //  8.39 MB  (end 142.6 MB)

  // Weight conversions (fp32 -> bf16), 1024 elems / block
  cvt_f32_bf16_kernel<<<3072, 256, 0, stream>>>(w_qkv,  wqkvb);
  cvt_f32_bf16_kernel<<<1024, 256, 0, stream>>>(w_proj, wprojb);
  cvt_f32_bf16_kernel<<<4096, 256, 0, stream>>>(w_fc1,  wfc1b);
  cvt_f32_bf16_kernel<<<4096, 256, 0, stream>>>(w_fc2,  wfc2b);

  // LN1 -> xn (bf16)
  ln_kernel<<<8192, 256, 0, stream>>>(x, ln1_g, ln1_b, xn);

  // QKV GEMM: [8192,1024] x [1024,3072] -> bf16
  gemm_bf16_kernel<false, false, false, true><<<dim3(24, 64), 256, 0, stream>>>(
      xn, wqkvb, nullptr, qkvb, nullptr, nullptr, 8192, 3072, 1024);

  // Attention: 8 batches * 16 heads * 8 q-blocks
  attn_kernel<<<1024, 256, 0, stream>>>(qkvb, attnb);

  // Proj GEMM + bias + residual(x) -> x1 (fp32)
  gemm_bf16_kernel<true, false, true, false><<<dim3(8, 64), 256, 0, stream>>>(
      attnb, wprojb, x1, nullptr, b_proj, x, 8192, 1024, 1024);

  // LN2 -> xn (bf16, buffer reuse)
  ln_kernel<<<8192, 256, 0, stream>>>(x1, ln2_g, ln2_b, xn);

  // FC1 GEMM + bias + exact GELU -> h (bf16)
  gemm_bf16_kernel<true, true, false, true><<<dim3(32, 64), 256, 0, stream>>>(
      xn, wfc1b, nullptr, hbuf, b_fc1, nullptr, 8192, 4096, 1024);

  // FC2 GEMM + bias + residual(x1) -> out (fp32)
  gemm_bf16_kernel<true, false, true, false><<<dim3(8, 64), 256, 0, stream>>>(
      hbuf, wfc2b, out, nullptr, b_fc2, x1, 8192, 1024, 4096);
}
